// StackedRNN_7902739824855
// MI455X (gfx1250) — matrix operand order repaired and checked by reading, not compiled, not run on hardware
//
#include <hip/hip_runtime.h>

#define H_ 512
#define B_ 8
#define S_ 4096
#define L_ 4
#define M_ (B_*S_)
#define CHUNKS 64
#define TCH (S_/CHUNKS)

typedef __attribute__((ext_vector_type(16))) __bf16 v16bf;
typedef __attribute__((ext_vector_type(8)))  float  v8f;

struct alignas(16) BF8 { __bf16 h[8]; };

__device__ __forceinline__ unsigned short f2bf(float f) {
  unsigned u = __float_as_uint(f);
  u = u + 0x7FFFu + ((u >> 16) & 1u);     // round-to-nearest-even
  return (unsigned short)(u >> 16);
}

// ---------------- elementwise helpers ----------------

__global__ __launch_bounds__(256)
void k_f32_to_bf16(const float* __restrict__ in, unsigned short* __restrict__ out, int n) {
  int i = blockIdx.x * 256 + threadIdx.x;
  if (i < n) out[i] = f2bf(in[i]);
}

// out[b, S-1-s, h] = in[b, s, h]
__global__ __launch_bounds__(256)
void k_flip(const float* __restrict__ in, float* __restrict__ out) {
  int i = blockIdx.x * 256 + threadIdx.x;
  if (i >= M_ * H_) return;
  int h = i % H_;
  int bs = i / H_;
  int s = bs % S_;
  int b = bs / S_;
  out[((size_t)(b * S_ + (S_ - 1 - s))) * H_ + h] = in[i];
}

// Wcat[r,k] (bf16), r in [0,1024): rows 0..511 = Wz, 512..1023 = Wh
__global__ __launch_bounds__(256)
void k_wcat(const float* __restrict__ Wz, const float* __restrict__ Wh,
            unsigned short* __restrict__ out) {
  int i = blockIdx.x * 256 + threadIdx.x;
  if (i >= 1024 * H_) return;
  int k = i % H_;
  int r = i / H_;
  float v = (r < H_) ? Wz[r * H_ + k] : Wh[(r - H_) * H_ + k];
  out[i] = f2bf(v);
}

__global__ __launch_bounds__(256)
void k_bcat(const float* __restrict__ bz, const float* __restrict__ bh,
            float* __restrict__ out) {
  int i = blockIdx.x * 256 + threadIdx.x;
  if (i >= 1024) return;
  out[i] = (i < H_) ? bz[i] : bh[i - H_];
}

// pack act (B,S,H) f32 into concat bf16 (B,S,2H), optionally sequence-flipped
__global__ __launch_bounds__(256)
void k_concat(const float* __restrict__ act, unsigned short* __restrict__ out,
              int half, int flip) {
  int i = blockIdx.x * 256 + threadIdx.x;
  if (i >= M_ * H_) return;
  int h = i % H_;
  int bs = i / H_;
  int s = bs % S_;
  int b = bs / S_;
  int so = flip ? (S_ - 1 - s) : s;
  out[((size_t)(b * S_ + so)) * 1024 + half * H_ + h] = f2bf(act[i]);
}

// ---------------- WMMA GEMM: C[m,n] = act( sum_k A[m,k]*W[n,k] + bias[n] ) --
// A: (M,K) bf16 row-major, W: (N,K) bf16 row-major, C: (M,N) f32
// act==1: n<512 -> sigmoid (z gate), n>=512 -> tanh (h~ gate)

__global__ __launch_bounds__(128)
void k_wmma_gemm(const unsigned short* __restrict__ Abf,
                 const unsigned short* __restrict__ Wbf,
                 const float* __restrict__ bias,
                 float* __restrict__ C,
                 int M, int N, int K, int act) {
  const __bf16* A = (const __bf16*)Abf;
  const __bf16* W = (const __bf16*)Wbf;
  const int lane = threadIdx.x & 31;
  const int wave = threadIdx.x >> 5;
  const int m0 = blockIdx.y * 128 + (wave >> 1) * 64;
  const int n0 = blockIdx.x * 128 + (wave & 1) * 64;
  const int lm = lane & 15;   // row (A) / column (B) within tile
  const int lh = lane >> 4;   // half-wave selector for K chunks

  v8f acc[4][4];
#pragma unroll
  for (int i = 0; i < 4; i++)
#pragma unroll
    for (int j = 0; j < 4; j++)
      acc[i][j] = {};

  for (int k0 = 0; k0 < K; k0 += 32) {
    v16bf af[4], bfr[4];
    // A fragments: 16x32 bf16. Lane holds row m=lm; K = k0+8*lh..+7 and +16.
#pragma unroll
    for (int i = 0; i < 4; i++) {
      const __bf16* ap = A + (size_t)(m0 + 16 * i + lm) * K + k0 + lh * 8;
      BF8 lo = *(const BF8*)ap;
      BF8 hi = *(const BF8*)(ap + 16);
#pragma unroll
      for (int j = 0; j < 8; j++) { af[i][j] = lo.h[j]; af[i][8 + j] = hi.h[j]; }
    }
    // B fragments: 32x16. Lane holds column n=lm; K = k0+8*lh..+7 and +16.
#pragma unroll
    for (int j = 0; j < 4; j++) {
      const __bf16* wp = W + (size_t)(n0 + 16 * j + lm) * K + k0 + lh * 8;
      BF8 lo = *(const BF8*)wp;
      BF8 hi = *(const BF8*)(wp + 16);
#pragma unroll
      for (int t = 0; t < 8; t++) { bfr[j][t] = lo.h[t]; bfr[j][8 + t] = hi.h[t]; }
    }
#pragma unroll
    for (int i = 0; i < 4; i++)
#pragma unroll
      for (int j = 0; j < 4; j++)
        acc[i][j] = __builtin_amdgcn_wmma_f32_16x16x32_bf16(
            false, af[i], false, bfr[j], (short)0, acc[i][j], false, false);
  }

  // Epilogue: D layout = VGPR r -> M = r + 8*lh, N = lm.
  // Gate type (sigmoid vs tanh) is uniform per 16-wide tile -> scalar select.
#pragma unroll
  for (int j = 0; j < 4; j++) {
    const int n = n0 + 16 * j + lm;
    const float bv = bias[n];
    const bool zgate = (n0 + 16 * j) < H_;
#pragma unroll
    for (int i = 0; i < 4; i++)
#pragma unroll
      for (int r = 0; r < 8; r++) {
        int m = m0 + 16 * i + lh * 8 + r;
        float y = acc[i][j][r] + bv;
        if (act == 1) {
          if (zgate) y = 1.f / (1.f + __expf(-y));                 // sigmoid
          else       y = 2.f / (1.f + __expf(-2.f * y)) - 1.f;     // tanh
        }
        C[(size_t)m * N + n] = y;
      }
  }
}

// ---------------- chunked linear-recurrence scan ----------------
// Gates G: (M, 1024) f32, z at [.., h], h~ at [.., 512+h]
// a = clamp(1-z, 1e-8, 1-1e-8); b = z*h~; h_t = a_t h_{t-1} + b_t, h_{-1}=0

__global__ __launch_bounds__(512)
void k_scan1(const float* __restrict__ G, float* __restrict__ P, float* __restrict__ Sv) {
  int h = threadIdx.x, c = blockIdx.x, b = blockIdx.y;
  const float* g = G + ((size_t)(b * S_ + c * TCH)) * 1024;
  float p = 1.f, s = 0.f;
  for (int t = 0; t < TCH; t++) {
    if (t + 8 < TCH) {
      __builtin_prefetch(g + (size_t)(t + 8) * 1024 + h, 0, 0);
      __builtin_prefetch(g + (size_t)(t + 8) * 1024 + H_ + h, 0, 0);
    }
    float z  = g[(size_t)t * 1024 + h];
    float ht = g[(size_t)t * 1024 + H_ + h];
    float a = fminf(fmaxf(1.f - z, 1e-8f), 1.f - 1e-8f);
    s = a * s + z * ht;
    p = a * p;
  }
  size_t o = ((size_t)(b * CHUNKS + c)) * H_ + h;
  P[o] = p; Sv[o] = s;
}

__global__ __launch_bounds__(512)
void k_scan2(const float* __restrict__ P, const float* __restrict__ Sv,
             float* __restrict__ E) {
  int h = threadIdx.x, b = blockIdx.x;
  float carry = 0.f;
  for (int c = 0; c < CHUNKS; c++) {
    size_t o = ((size_t)(b * CHUNKS + c)) * H_ + h;
    carry = P[o] * carry + Sv[o];
    E[o] = carry;   // state at end of chunk c
  }
}

// Pass 3 fused with residual + LayerNorm + bf16 emit for the next layer's GEMM.
// Block = (chunk c, batch b), 512 threads = full H row per timestep.
__global__ __launch_bounds__(512)
void k_scan3ln(const float* __restrict__ G, const float* __restrict__ E,
               const float* __restrict__ resid, int use_res,
               const float* __restrict__ lng, const float* __restrict__ lnb,
               float* __restrict__ outf, unsigned short* __restrict__ outbf) {
  __shared__ float psum[2][16];
  __shared__ float psq[2][16];
  const int h = threadIdx.x, c = blockIdx.x, b = blockIdx.y;
  const int lane = h & 31, wv = h >> 5;
  float s = (c == 0) ? 0.f : E[((size_t)(b * CHUNKS + c - 1)) * H_ + h];
  const size_t base = (size_t)(b * S_ + c * TCH);
  const float* g  = G + base * 1024;
  const float* rp = resid + base * H_;
  float* of = outf + base * H_;
  unsigned short* ob = outbf + base * H_;
  const float gh = lng[h], bh = lnb[h];

  for (int t = 0; t < TCH; t++) {
    if (t + 8 < TCH) {
      __builtin_prefetch(g + (size_t)(t + 8) * 1024 + h, 0, 0);
      __builtin_prefetch(g + (size_t)(t + 8) * 1024 + H_ + h, 0, 0);
    }
    float z  = g[(size_t)t * 1024 + h];
    float ht = g[(size_t)t * 1024 + H_ + h];
    float a = fminf(fmaxf(1.f - z, 1e-8f), 1.f - 1e-8f);
    s = a * s + z * ht;                                   // scan state
    float v = s + (use_res ? rp[(size_t)t * H_ + h] : 0.f);

    // cross-wave LN reduction: wave32 shuffle, then 16 partials via LDS
    float w1 = v, w2 = v * v;
#pragma unroll
    for (int d = 16; d > 0; d >>= 1) {
      w1 += __shfl_xor(w1, d, 32);
      w2 += __shfl_xor(w2, d, 32);
    }
    if (lane == 0) { psum[t & 1][wv] = w1; psq[t & 1][wv] = w2; }
    __syncthreads();
    float tot = 0.f, tot2 = 0.f;
#pragma unroll
    for (int i = 0; i < 16; i++) { tot += psum[t & 1][i]; tot2 += psq[t & 1][i]; }
    float mu  = tot * (1.f / H_);
    float var = tot2 * (1.f / H_) - mu * mu;
    float rs  = rsqrtf(fmaxf(var, 0.f) + 1e-5f);
    float o = (v - mu) * rs * gh + bh;
    of[(size_t)t * H_ + h] = o;
    ob[(size_t)t * H_ + h] = f2bf(o);
  }
}

// ---------------- host orchestration ----------------

extern "C" void kernel_launch(void* const* d_in, const int* in_sizes, int n_in,
                              void* d_out, int out_size, void* d_ws, size_t ws_size,
                              hipStream_t stream) {
  const float* x        = (const float*)d_in[0];
  const float* Wz_f     = (const float*)d_in[1];
  const float* bz_f     = (const float*)d_in[2];
  const float* Wh_f     = (const float*)d_in[3];
  const float* bh_f     = (const float*)d_in[4];
  const float* Wz_b     = (const float*)d_in[5];
  const float* bz_b     = (const float*)d_in[6];
  const float* Wh_b     = (const float*)d_in[7];
  const float* bh_b     = (const float*)d_in[8];
  const float* fusion_w = (const float*)d_in[9];
  const float* fusion_b = (const float*)d_in[10];
  const float* ln_g     = (const float*)d_in[11];
  const float* ln_b     = (const float*)d_in[12];

  char* ws = (char*)d_ws;
  size_t off = 0;
  auto alloc = [&](size_t bytes) -> char* {
    char* p = ws + off;
    off += (bytes + 255) & ~(size_t)255;
    return p;
  };
  float*          gates  = (float*)alloc((size_t)M_ * 1024 * 4);  // 128 MB
  float*          actA   = (float*)alloc((size_t)M_ * H_ * 4);    // 64 MB
  float*          actB   = (float*)alloc((size_t)M_ * H_ * 4);    // 64 MB
  unsigned short* abf    = (unsigned short*)alloc((size_t)M_ * H_ * 2);   // 32 MB
  unsigned short* catbuf = (unsigned short*)alloc((size_t)M_ * 1024 * 2); // 64 MB
  unsigned short* wcat   = (unsigned short*)alloc((size_t)1024 * H_ * 2); // 1 MB
  float*          bcat   = (float*)alloc(1024 * 4);
  float*          P      = (float*)alloc((size_t)B_ * CHUNKS * H_ * 4);
  float*          Sv     = (float*)alloc((size_t)B_ * CHUNKS * H_ * 4);
  float*          E      = (float*)alloc((size_t)B_ * CHUNKS * H_ * 4);
  (void)ws_size; (void)in_sizes; (void)n_in; (void)out_size;

  const int nAct = M_ * H_;
  dim3 eb(256), eg((nAct + 255) / 256);
  dim3 wg((1024 * H_ + 255) / 256);
  dim3 sg(CHUNKS, B_);

  for (int dir = 0; dir < 2; dir++) {
    const float* Wz = dir ? Wz_b : Wz_f;
    const float* bz = dir ? bz_b : bz_f;
    const float* Wh = dir ? Wh_b : Wh_f;
    const float* bh = dir ? bh_b : bh_f;

    const float* cur;
    if (dir == 0) {
      cur = x;
    } else {
      k_flip<<<eg, eb, 0, stream>>>(x, actA);
      cur = actA;
    }
    float* outbuf = (dir == 0) ? actA : actB;

    for (int l = 0; l < L_; l++) {
      k_wcat<<<wg, eb, 0, stream>>>(Wz + (size_t)l * H_ * H_, Wh + (size_t)l * H_ * H_, wcat);
      k_bcat<<<4, eb, 0, stream>>>(bz + l * H_, bh + l * H_, bcat);
      if (l == 0)  // layers > 0 get bf16 activations directly from k_scan3ln
        k_f32_to_bf16<<<eg, eb, 0, stream>>>(cur, abf, nAct);

      dim3 gg(1024 / 128, M_ / 128);
      k_wmma_gemm<<<gg, 128, 0, stream>>>(abf, wcat, bcat, gates, M_, 1024, H_, 1);

      k_scan1<<<sg, 512, 0, stream>>>(gates, P, Sv);
      k_scan2<<<B_, 512, 0, stream>>>(P, Sv, E);
      // fused: scan + residual + LayerNorm + f32 act + bf16 act for next GEMM
      k_scan3ln<<<sg, 512, 0, stream>>>(gates, E, cur, (l > 0) ? 1 : 0,
                                        ln_g, ln_b, outbuf, abf);
      cur = outbuf;
      outbuf = (outbuf == actA) ? actB : actA;
    }
    // pack final activations of this direction into concat bf16 buffer
    k_concat<<<eg, eb, 0, stream>>>(cur, catbuf, dir, dir);
  }

  // fusion: out = concat(B,S,2H) @ fusion_w(512,1024)^T + fusion_b
  k_f32_to_bf16<<<(512 * 1024 + 255) / 256, eb, 0, stream>>>(fusion_w, wcat, 512 * 1024);
  dim3 fg(512 / 128, M_ / 128);
  k_wmma_gemm<<<fg, 128, 0, stream>>>(catbuf, wcat, fusion_b, (float*)d_out,
                                      M_, 512, 1024, 0);
}
